// LSTMDiscriminator_16183436771659
// MI455X (gfx1250) — matrix operand order repaired and checked by reading, not compiled, not run on hardware
//
#include <hip/hip_runtime.h>
#include <hip/hip_bf16.h>

typedef __attribute__((ext_vector_type(16))) __bf16        v16bf;
typedef __attribute__((ext_vector_type(8)))  float         v8f;
typedef __attribute__((ext_vector_type(4)))  unsigned int  u32x4;  // builtin 16B vector (AS-agnostic)

#define AS1 __attribute__((address_space(1)))
typedef const unsigned short AS1* gwp;   // global (AS1) bf16-weight pointer
typedef const u32x4 AS1*          gq4p;  // global (AS1) 16-byte chunk pointer

#define L_SEQ 512
#define BATCH 1024
#define XD    128
#define HD    256
#define GN    1024   // 4*HD gate columns
#define BT    32     // batch tile per workgroup
#define NWG   (BATCH / BT)   // 32 workgroups
#define GST   1024   // gates LDS row stride (f32 elems)

union Frag16 { v16bf v; u32x4 q[2]; };

__device__ __forceinline__ unsigned short f2bf(float f) {
  union { float f; unsigned u; } x; x.f = f;
  unsigned r = x.u + 0x7FFFu + ((x.u >> 16) & 1u);   // round-to-nearest-even
  return (unsigned short)(r >> 16);
}
__device__ __forceinline__ float bf2f(unsigned short h) {
  union { unsigned u; float f; } x; x.u = ((unsigned)h) << 16;
  return x.f;
}
__device__ __forceinline__ float sigm(float x) { return 1.0f / (1.0f + __expf(-x)); }
__device__ __forceinline__ float leaky(float z) { return (z >= 0.0f) ? z : 0.2f * z; }

// Opaque copy of a GLOBAL (AS1) pointer: forbids hoisting loads out of the
// t-loop while keeping the global_load path (no FLAT, no DScnt coupling).
__device__ __forceinline__ gwp launder_g(gwp p) {
  asm volatile("" : "+s"(p));
  return p;
}

// A fragment: 16x32 bf16 (MxK) from LDS row-major buffer, row stride = `stride` elems.
// Lanes 0-15: row m, K = kc*32 + {0..7, 16..23}; lanes 16-31: K = kc*32 + {8..15, 24..31}.
__device__ __forceinline__ v16bf load_a(const unsigned short* buf, int mbase, int stride, int kc) {
  const int lane = threadIdx.x & 31;
  const int m  = mbase + (lane & 15);
  const int kb = kc * 32 + ((lane & 16) ? 8 : 0);
  const unsigned short* p = buf + m * stride + kb;
  Frag16 f;
  f.q[0] = *(const u32x4*)(p);        // K kb..kb+7
  f.q[1] = *(const u32x4*)(p + 16);   // K kb+16..kb+23
  return f.v;
}

// B fragment: 32x16 bf16 (KxN) from global row-major W[N][ktot] (bf16).
// Lane n = n0 + lane%16; lanes 0-15: K = kc*32 + {0..7, 16..23}; lanes 16-31: K = kc*32 + {8..15, 24..31}.
__device__ __forceinline__ v16bf load_b(gwp W, int ktot, int n0, int kc) {
  const int lane = threadIdx.x & 31;
  const int n  = n0 + (lane & 15);
  const int kb = kc * 32 + ((lane & 16) ? 8 : 0);
  gwp p = W + (size_t)n * ktot + kb;
  Frag16 f;
  f.q[0] = *(gq4p)(p);        // K kb..kb+7
  f.q[1] = *(gq4p)(p + 16);   // K kb+16..kb+23
  return f.v;
}

__device__ __forceinline__ v8f wmma_bf16(v16bf a, v16bf b, v8f c) {
  return __builtin_amdgcn_wmma_f32_16x16x32_bf16(false, a, false, b, (short)0, c, false, false);
}

// ---------------- prep: convert weights to bf16 in L2-resident workspace ----------------
__global__ void prep_weights(const float* __restrict__ Wih, const float* __restrict__ Whh,
                             const float* __restrict__ bih, const float* __restrict__ bhh,
                             const float* __restrict__ W0,  const float* __restrict__ W1,
                             unsigned short* gWih, unsigned short* gWhh,
                             unsigned short* gW0,  unsigned short* gW1, float* biasc) {
  const int tid = blockIdx.x * blockDim.x + threadIdx.x;
  const int n = gridDim.x * blockDim.x;
  for (int i = tid; i < GN * XD; i += n) gWih[i] = f2bf(Wih[i]);
  for (int i = tid; i < GN * HD; i += n) gWhh[i] = f2bf(Whh[i]);
  for (int i = tid; i < HD * HD; i += n) { gW0[i] = f2bf(W0[i]); gW1[i] = f2bf(W1[i]); }
  for (int i = tid; i < GN; i += n) biasc[i] = bih[i] + bhh[i];
}

// ---------------- main persistent LSTM + MLP head kernel ----------------
__global__ void __launch_bounds__(256, 1)
lstm_disc(const float* __restrict__ x,
          const unsigned short* __restrict__ gWih0,
          const unsigned short* __restrict__ gWhh0,
          const unsigned short* __restrict__ gW00,
          const unsigned short* __restrict__ gW10,
          const float* __restrict__ biasc,
          const float* __restrict__ b0,
          const float* __restrict__ b1,
          const float* __restrict__ W2,
          const float* __restrict__ b2,
          float* __restrict__ out) {
  extern __shared__ char smem[];
  float*          gates = (float*)smem;                                   // 32x1024 f32 = 128KB
  unsigned short* xtile = (unsigned short*)(smem + BT * GST * 4);         // 32x128 bf16 = 8KB
  unsigned short* hbuf  = (unsigned short*)(smem + BT * GST * 4 + BT * XD * 2); // 32x256 bf16 = 16KB
  float*          cbuf  = (float*)(smem + BT * GST * 4 + BT * XD * 2 + BT * HD * 2); // 32x256 f32 = 32KB
  unsigned short* ob0   = (unsigned short*)smem;                          // reuse gates region
  unsigned short* ob1   = (unsigned short*)(smem + BT * HD * 2);

  const int tid  = threadIdx.x;
  const int lane = tid & 31;
  const int wave = tid >> 5;
  const int wg   = blockIdx.x;

  // global (AS1) views of the bf16 weights
  gwp wih_g = (gwp)(unsigned long long)gWih0;
  gwp whh_g = (gwp)(unsigned long long)gWhh0;
  gwp w0_g  = (gwp)(unsigned long long)gW00;
  gwp w1_g  = (gwp)(unsigned long long)gW10;

  // init h = 0, c = 0
  for (int i = tid; i < BT * HD; i += 256) { hbuf[i] = 0; cbuf[i] = 0.0f; }
  __syncthreads();

  const int n0w = wave * 128;  // this wave's gate-column base (8 N-tiles)
  const int j0  = wave * 32;   // this wave's h-column slice for activations

  // per-wave gate bias, one column per lane (C/D layout), kept in registers
  float bc[8];
  #pragma unroll
  for (int nt = 0; nt < 8; ++nt) bc[nt] = biasc[n0w + nt * 16 + (lane & 15)];

  for (int t = 0; t < L_SEQ; ++t) {
    // opaque weight pointers: loads cannot be hoisted out of the t-loop
    gwp wih = launder_g(wih_g);
    gwp whh = launder_g(whh_g);

    // ---- stage x_t tile (32x128) to LDS as bf16 ----
    const float* xt = x + ((size_t)t * BATCH + (size_t)wg * BT) * XD;
    for (int v = tid; v < BT * XD / 4; v += 256) {
      float4 f = ((const float4*)xt)[v];
      ushort4 h4;
      h4.x = f2bf(f.x); h4.y = f2bf(f.y); h4.z = f2bf(f.z); h4.w = f2bf(f.w);
      ((ushort4*)xtile)[v] = h4;
    }
    // prefetch next step's x tile (sequential 16KB stream; 256 threads x 64B)
    if (t + 1 < L_SEQ) {
      const char* nxt = (const char*)(x + ((size_t)(t + 1) * BATCH + (size_t)wg * BT) * XD);
      __builtin_prefetch(nxt + tid * 64, 0, 3);
    }
    __syncthreads();

    // ---- gates = x_t @ Wih^T + h @ Whh^T + (b_ih + b_hh) ----
    v8f acc[2][8];
    #pragma unroll
    for (int nt = 0; nt < 8; ++nt)
      #pragma unroll
      for (int mt = 0; mt < 2; ++mt)
        #pragma unroll
        for (int r = 0; r < 8; ++r) acc[mt][nt][r] = bc[nt];

    #pragma unroll
    for (int kc = 0; kc < XD / 32; ++kc) {         // input contribution, K=128
      // batch-issue all 8 B fragments (16 global_load_b128 in flight)
      v16bf bfr[8];
      #pragma unroll
      for (int nt = 0; nt < 8; ++nt) bfr[nt] = load_b(wih, XD, n0w + nt * 16, kc);
      v16bf a0 = load_a(xtile, 0, XD, kc);
      v16bf a1 = load_a(xtile, 16, XD, kc);
      #pragma unroll
      for (int nt = 0; nt < 8; ++nt) {
        acc[0][nt] = wmma_bf16(a0, bfr[nt], acc[0][nt]);
        acc[1][nt] = wmma_bf16(a1, bfr[nt], acc[1][nt]);
      }
    }
    #pragma unroll
    for (int kc = 0; kc < HD / 32; ++kc) {         // recurrent contribution, K=256
      v16bf bfr[8];
      #pragma unroll
      for (int nt = 0; nt < 8; ++nt) bfr[nt] = load_b(whh, HD, n0w + nt * 16, kc);
      v16bf a0 = load_a(hbuf, 0, HD, kc);
      v16bf a1 = load_a(hbuf, 16, HD, kc);
      #pragma unroll
      for (int nt = 0; nt < 8; ++nt) {
        acc[0][nt] = wmma_bf16(a0, bfr[nt], acc[0][nt]);
        acc[1][nt] = wmma_bf16(a1, bfr[nt], acc[1][nt]);
      }
    }
    // store gate slice to LDS (C/D layout: fixed column per lane, 8 rows per VGPR set)
    #pragma unroll
    for (int nt = 0; nt < 8; ++nt) {
      const int col = n0w + nt * 16 + (lane & 15);
      #pragma unroll
      for (int mt = 0; mt < 2; ++mt) {
        const int rb = mt * 16 + ((lane & 16) ? 8 : 0);
        #pragma unroll
        for (int r = 0; r < 8; ++r)
          gates[(rb + r) * GST + col] = acc[mt][nt][r];
      }
    }
    __syncthreads();

    // ---- pointwise LSTM cell update: wave handles h-cols [j0, j0+32) ----
    const int j = j0 + lane;
    for (int b = 0; b < BT; ++b) {
      float gi = gates[b * GST + j];
      float gf = gates[b * GST + j + 256];
      float gg = gates[b * GST + j + 512];
      float go = gates[b * GST + j + 768];
      float c  = cbuf[b * HD + j];
      float cn = sigm(gf) * c + sigm(gi) * tanhf(gg);
      float hn = sigm(go) * tanhf(cn);
      cbuf[b * HD + j] = cn;
      hbuf[b * HD + j] = f2bf(hn);
    }
    __syncthreads();
  }

  // ---- head layer 0: out0 = leaky(h @ W0^T + b0), 32x256, WMMA ----
  {
    v8f acc[2][2];
    #pragma unroll
    for (int nt = 0; nt < 2; ++nt) {
      float bb = b0[wave * 32 + nt * 16 + (lane & 15)];
      #pragma unroll
      for (int mt = 0; mt < 2; ++mt)
        #pragma unroll
        for (int r = 0; r < 8; ++r) acc[mt][nt][r] = bb;
    }
    #pragma unroll
    for (int kc = 0; kc < HD / 32; ++kc) {
      v16bf bfr[2];
      #pragma unroll
      for (int nt = 0; nt < 2; ++nt) bfr[nt] = load_b(w0_g, HD, wave * 32 + nt * 16, kc);
      v16bf a0 = load_a(hbuf, 0, HD, kc);
      v16bf a1 = load_a(hbuf, 16, HD, kc);
      #pragma unroll
      for (int nt = 0; nt < 2; ++nt) {
        acc[0][nt] = wmma_bf16(a0, bfr[nt], acc[0][nt]);
        acc[1][nt] = wmma_bf16(a1, bfr[nt], acc[1][nt]);
      }
    }
    #pragma unroll
    for (int nt = 0; nt < 2; ++nt) {
      const int col = wave * 32 + nt * 16 + (lane & 15);
      #pragma unroll
      for (int mt = 0; mt < 2; ++mt) {
        const int rb = mt * 16 + ((lane & 16) ? 8 : 0);
        #pragma unroll
        for (int r = 0; r < 8; ++r)
          ob0[(rb + r) * HD + col] = f2bf(leaky(acc[mt][nt][r]));
      }
    }
  }
  __syncthreads();

  // ---- head layer 1: out1 = leaky(out0 @ W1^T + b1) ----
  {
    v8f acc[2][2];
    #pragma unroll
    for (int nt = 0; nt < 2; ++nt) {
      float bb = b1[wave * 32 + nt * 16 + (lane & 15)];
      #pragma unroll
      for (int mt = 0; mt < 2; ++mt)
        #pragma unroll
        for (int r = 0; r < 8; ++r) acc[mt][nt][r] = bb;
    }
    #pragma unroll
    for (int kc = 0; kc < HD / 32; ++kc) {
      v16bf bfr[2];
      #pragma unroll
      for (int nt = 0; nt < 2; ++nt) bfr[nt] = load_b(w1_g, HD, wave * 32 + nt * 16, kc);
      v16bf a0 = load_a(ob0, 0, HD, kc);
      v16bf a1 = load_a(ob0, 16, HD, kc);
      #pragma unroll
      for (int nt = 0; nt < 2; ++nt) {
        acc[0][nt] = wmma_bf16(a0, bfr[nt], acc[0][nt]);
        acc[1][nt] = wmma_bf16(a1, bfr[nt], acc[1][nt]);
      }
    }
    #pragma unroll
    for (int nt = 0; nt < 2; ++nt) {
      const int col = wave * 32 + nt * 16 + (lane & 15);
      #pragma unroll
      for (int mt = 0; mt < 2; ++mt) {
        const int rb = mt * 16 + ((lane & 16) ? 8 : 0);
        #pragma unroll
        for (int r = 0; r < 8; ++r)
          ob1[(rb + r) * HD + col] = f2bf(leaky(acc[mt][nt][r]));
      }
    }
  }
  __syncthreads();

  // ---- final projection: out[b] = out1[b,:] . W2 + b2 ----
  if (tid < BT) {
    float s = 0.0f;
    for (int k = 0; k < HD; ++k) s += bf2f(ob1[tid * HD + k]) * W2[k];
    out[wg * BT + tid] = s + b2[0];
  }
}

extern "C" void kernel_launch(void* const* d_in, const int* in_sizes, int n_in,
                              void* d_out, int out_size, void* d_ws, size_t ws_size,
                              hipStream_t stream) {
  const float* x    = (const float*)d_in[0];
  const float* W_ih = (const float*)d_in[1];
  const float* W_hh = (const float*)d_in[2];
  const float* b_ih = (const float*)d_in[3];
  const float* b_hh = (const float*)d_in[4];
  const float* W0   = (const float*)d_in[5];
  const float* b0   = (const float*)d_in[6];
  const float* W1   = (const float*)d_in[7];
  const float* b1   = (const float*)d_in[8];
  const float* W2   = (const float*)d_in[9];
  const float* b2   = (const float*)d_in[10];
  float* out = (float*)d_out;

  // workspace layout: bf16 weights + combined gate bias (~1.03 MB, L2-resident)
  unsigned short* gWih = (unsigned short*)d_ws;                // 1024*128
  unsigned short* gWhh = gWih + GN * XD;                       // 1024*256
  unsigned short* gW0  = gWhh + GN * HD;                       // 256*256
  unsigned short* gW1  = gW0 + HD * HD;                        // 256*256
  float*          bsc  = (float*)(gW1 + HD * HD);              // 1024

  prep_weights<<<256, 256, 0, stream>>>(W_ih, W_hh, b_ih, b_hh, W0, W1,
                                        gWih, gWhh, gW0, gW1, bsc);

  const size_t lds_bytes = (size_t)BT * GST * 4 + (size_t)BT * XD * 2 +
                           (size_t)BT * HD * 2 + (size_t)BT * HD * 4;  // 188416
  (void)hipFuncSetAttribute((const void*)lstm_disc,
                            hipFuncAttributeMaxDynamicSharedMemorySize, (int)lds_bytes);
  lstm_disc<<<NWG, 256, lds_bytes, stream>>>(x, gWih, gWhh, gW0, gW1, bsc,
                                             b0, b1, W2, b2, out);
}
